// JKNet_54984171323612
// MI455X (gfx1250) — matrix operand order repaired and checked by reading, compiled, not run on hardware
//
#include <hip/hip_runtime.h>
#include <hip/hip_bf16.h>

typedef __attribute__((ext_vector_type(16))) _Float16 v16h;
typedef __attribute__((ext_vector_type(8)))  _Float16 v8h;
typedef __attribute__((ext_vector_type(8)))  float    v8f;

#define IN_DIM 128
#define HID    128
#define OUTD   64

// ---------------- elementwise helpers ----------------

__global__ void zero_kernel(float* __restrict__ p, long long n) {
    long long i = (long long)blockIdx.x * blockDim.x + threadIdx.x;
    long long s = (long long)gridDim.x * blockDim.x;
    for (; i < n; i += s) p[i] = 0.0f;
}

__global__ void degree_kernel(const int* __restrict__ src, const int* __restrict__ dst,
                              float* __restrict__ deg_out, float* __restrict__ deg_in, int E) {
    long long i = (long long)blockIdx.x * blockDim.x + threadIdx.x;
    long long s = (long long)gridDim.x * blockDim.x;
    for (; i < E; i += s) {
        atomicAdd(&deg_out[src[i]], 1.0f);
        atomicAdd(&deg_in[dst[i]], 1.0f);
    }
}

__global__ void rsqrt_kernel(float* __restrict__ p, long long n) {
    long long i = (long long)blockIdx.x * blockDim.x + threadIdx.x;
    long long s = (long long)gridDim.x * blockDim.x;
    for (; i < n; i += s) {
        float v = p[i];
        p[i] = rsqrtf(fmaxf(v, 1.0f));
    }
}

__global__ void cvt_f16_kernel(const float* __restrict__ w, _Float16* __restrict__ o, int n) {
    int i = blockIdx.x * blockDim.x + threadIdx.x;
    if (i < n) o[i] = (_Float16)w[i];
}

// h = relu(agg * inv_in[row] + b[col]), C fixed at 128
__global__ void post_kernel(const float* __restrict__ agg, const float* __restrict__ inv_in,
                            const float* __restrict__ b, float* __restrict__ h, long long total) {
    long long i = (long long)blockIdx.x * blockDim.x + threadIdx.x;
    long long s = (long long)gridDim.x * blockDim.x;
    for (; i < total; i += s) {
        float v = agg[i] * inv_in[i >> 7] + b[(int)i & 127];
        h[i] = fmaxf(v, 0.0f);
    }
}

// ---------------- edge scatter-add (SpMM copy_u/sum) ----------------
// One wave covers one edge's 128 channels (4 floats/lane, coalesced 512B gather).
__global__ void scatter_add_kernel(const float* __restrict__ T, const int* __restrict__ src,
                                   const int* __restrict__ dst, float* __restrict__ out,
                                   long long total, int cqShift, int C, int ldOut, int colOff) {
    long long i = (long long)blockIdx.x * blockDim.x + threadIdx.x;
    long long s = (long long)gridDim.x * blockDim.x;
    const int mask = (1 << cqShift) - 1;
    for (; i < total; i += s) {
        int e = (int)(i >> cqShift);
        int c = ((int)i & mask) << 2;
        int sn = src[e], dn = dst[e];
        const float4 v = *reinterpret_cast<const float4*>(T + (size_t)sn * C + c);
        float* o = out + (size_t)dn * ldOut + colOff + c;
        atomicAdd(o + 0, v.x);
        atomicAdd(o + 1, v.y);
        atomicAdd(o + 2, v.z);
        atomicAdd(o + 3, v.w);
    }
}

// ---------------- WMMA strip GEMM ----------------
// Out[M, NT*16] = diag(rowscale?) * X[M,K] (f32) @ Wh[K, NT*16] (f16) + bias
// Block = 64 threads (2 waves). Each wave owns a 16-row strip across all NT
// column tiles: A fragment loaded/converted once per K-step, NT unrolled WMMAs
// with NT resident f32 accumulators. B staged in LDS (padded stride K+8 halves:
// 16B-aligned v8h loads, bank-conflict-free across the 16 column lanes).
// grid.x = M/32 (M = 100000 -> 3125 exactly, EXEC stays all-ones).
template <int NT>
__global__ void __launch_bounds__(64)
wmma_gemm_strip(const float* __restrict__ X, const _Float16* __restrict__ Wh,
                const float* __restrict__ rowscale, const float* __restrict__ bias,
                float* __restrict__ Out, int K) {
    constexpr int Kout = NT * 16;
    constexpr int koutShift = (NT == 8) ? 7 : 6;   // log2(Kout)
    extern __shared__ _Float16 ldsB[];
    const int ldsStride = K + 8;                   // halves; 16B-aligned rows, conflict-free

    // cooperative transpose-fill: ldsB[n*ldsStride + k] = Wh[k*Kout + n]
    for (int i = threadIdx.x; i < K * Kout; i += 64) {
        int k = i >> koutShift;
        int n = i & (Kout - 1);
        ldsB[n * ldsStride + k] = Wh[i];
    }
    __syncthreads();

    const int wave = threadIdx.x >> 5;
    const int lane = threadIdx.x & 31;
    const int half = lane >> 4;          // 0 or 1
    const int l16  = lane & 15;          // A: row-in-tile ; B/D: col-in-tile
    const int m0   = (blockIdx.x * 2 + wave) * 16;

    const int arow = m0 + l16;
    const float s = rowscale ? rowscale[arow] : 1.0f;

    v8f acc[NT];
#pragma unroll
    for (int t = 0; t < NT; ++t) acc[t] = (v8f){};

    for (int kk = 0; kk < K; kk += 32) {
        // A fragment: a[j] -> k = kk + half*8 + j ; a[8+j] -> k = kk + half*8 + 16 + j
        v16h a;
        const float* xr = X + (size_t)arow * K + kk + half * 8;
#pragma unroll
        for (int j = 0; j < 8; ++j) {
            a[j]     = (_Float16)(xr[j] * s);
            a[8 + j] = (_Float16)(xr[16 + j] * s);
        }
#pragma unroll
        for (int t = 0; t < NT; ++t) {
            // B fragment (transposed LDS): b[j] -> B[kk + half*8 + j][t*16 + l16] ; b[8+j] -> B[kk + half*8 + 16 + j][t*16 + l16]
            const _Float16* bp = ldsB + (size_t)(t * 16 + l16) * ldsStride + kk + half * 8;
            union { v16h v; v8h h[2]; } u;
            u.h[0] = *reinterpret_cast<const v8h*>(bp);
            u.h[1] = *reinterpret_cast<const v8h*>(bp + 16);
            acc[t] = __builtin_amdgcn_wmma_f32_16x16x32_f16(
                /*neg_a=*/false, a, /*neg_b=*/false, u.v,
                /*c_mod=*/(short)0, acc[t], /*reuse_a=*/false, /*reuse_b=*/false);
        }
    }

    // D layout: acc[t][r] holds row (r + 8*half), col t*16 + l16
#pragma unroll
    for (int t = 0; t < NT; ++t) {
        const float bv = bias ? bias[t * 16 + l16] : 0.0f;
#pragma unroll
        for (int r = 0; r < 8; ++r) {
            int row = m0 + r + half * 8;
            Out[(size_t)row * Kout + t * 16 + l16] = acc[t][r] + bv;
        }
    }
}

// ---------------- launch ----------------

static inline long long cdivll(long long a, long long b) { return (a + b - 1) / b; }

extern "C" void kernel_launch(void* const* d_in, const int* in_sizes, int n_in,
                              void* d_out, int out_size, void* d_ws, size_t ws_size,
                              hipStream_t stream) {
    const float* x    = (const float*)d_in[0];
    const int*   src  = (const int*)d_in[1];
    const int*   dst  = (const int*)d_in[2];
    const float* W1   = (const float*)d_in[3];
    const float* b1   = (const float*)d_in[4];
    const float* W2   = (const float*)d_in[5];
    const float* b2   = (const float*)d_in[6];
    const float* Wout = (const float*)d_in[7];
    const float* bout = (const float*)d_in[8];
    float* out = (float*)d_out;

    const long long N = in_sizes[0] / IN_DIM;   // 100000 (divisible by 32)
    const int       E = in_sizes[1];            // 1600000

    // workspace layout (floats)
    float* wsf     = (float*)d_ws;
    float* inv_out = wsf;                 // N
    float* inv_in  = wsf + N;             // N
    float* h1      = wsf + 2 * N;         // 128N
    float* h2      = h1 + 128 * N;        // 128N
    float* S       = h2 + 128 * N;        // 256N : tmp=S[0:128N], agg=S[128N:256N]; later agged[N,256]
    float* tmp     = S;
    float* agg     = S + 128 * N;
    _Float16* W1h  = (_Float16*)(S + 256 * N);
    _Float16* W2h  = W1h + IN_DIM * HID;
    _Float16* Woh  = W2h + HID * HID;

    const int TB = 256;
    const int stripBlocks = (int)(N / 32);       // 3125

    // LDS bytes for the strip GEMM: Kout*(K+8)*2
    const size_t lds128 = (size_t)HID * (IN_DIM + 8) * sizeof(_Float16);    // 34816
    const size_t lds64  = (size_t)OUTD * (2 * HID + 8) * sizeof(_Float16);  // 33792

    // 1) degrees -> inv_sqrt
    zero_kernel<<<(int)cdivll(2 * N, TB), TB, 0, stream>>>(inv_out, 2 * N);
    degree_kernel<<<(int)cdivll(E, TB), TB, 0, stream>>>(src, dst, inv_out, inv_in, E);
    rsqrt_kernel<<<(int)cdivll(2 * N, TB), TB, 0, stream>>>(inv_out, 2 * N);

    // 2) weights -> f16
    cvt_f16_kernel<<<(IN_DIM * HID + TB - 1) / TB, TB, 0, stream>>>(W1, W1h, IN_DIM * HID);
    cvt_f16_kernel<<<(HID * HID + TB - 1) / TB, TB, 0, stream>>>(W2, W2h, HID * HID);
    cvt_f16_kernel<<<(2 * HID * OUTD + TB - 1) / TB, TB, 0, stream>>>(Wout, Woh, 2 * HID * OUTD);

    const long long eTot128 = (long long)E * (HID / 4);  // edge-elems, 4 ch/thread
    const int eBlocks = (int)cdivll(eTot128, TB);

    // 3) conv1: tmp = (x * inv_out) @ W1 ; agg = scatter(tmp) ; h1 = relu(agg*inv_in + b1)
    wmma_gemm_strip<8><<<stripBlocks, 64, lds128, stream>>>(x, W1h, inv_out, nullptr, tmp, IN_DIM);
    zero_kernel<<<(int)cdivll(128 * N, TB), TB, 0, stream>>>(agg, 128 * N);
    scatter_add_kernel<<<eBlocks, TB, 0, stream>>>(tmp, src, dst, agg, eTot128, 5, HID, HID, 0);
    post_kernel<<<(int)cdivll(128 * N, TB), TB, 0, stream>>>(agg, inv_in, b1, h1, 128 * N);

    // 4) conv2
    wmma_gemm_strip<8><<<stripBlocks, 64, lds128, stream>>>(h1, W2h, inv_out, nullptr, tmp, HID);
    zero_kernel<<<(int)cdivll(128 * N, TB), TB, 0, stream>>>(agg, 128 * N);
    scatter_add_kernel<<<eBlocks, TB, 0, stream>>>(tmp, src, dst, agg, eTot128, 5, HID, HID, 0);
    post_kernel<<<(int)cdivll(128 * N, TB), TB, 0, stream>>>(agg, inv_in, b2, h2, 128 * N);

    // 5) JK concat + aggregate: S[N,256] = scatter([h1|h2])
    zero_kernel<<<(int)cdivll(256 * N, TB), TB, 0, stream>>>(S, 256 * N);
    scatter_add_kernel<<<eBlocks, TB, 0, stream>>>(h1, src, dst, S, eTot128, 5, HID, 2 * HID, 0);
    scatter_add_kernel<<<eBlocks, TB, 0, stream>>>(h2, src, dst, S, eTot128, 5, HID, 2 * HID, HID);

    // 6) out = S @ Wout + bout
    wmma_gemm_strip<4><<<stripBlocks, 64, lds64, stream>>>(S, Woh, nullptr, bout, out, 2 * HID);
}
